// DynamicsFunction_89206470738319
// MI455X (gfx1250) — matrix operand order repaired and checked by reading, not compiled, not run on hardware
//
#include <hip/hip_runtime.h>
#include <hip/hip_bf16.h>
#include <math.h>

// ---------------------------------------------------------------------------
// Neural CDE (RK4 over cubic-spline control path) as a single persistent
// kernel on gfx1250. 32 blocks x 256 threads (8 wave32 each). Grid-wide
// sense barrier between RK stages. Both GEMMs per stage run on
// v_wmma_f32_16x16x32_bf16 with fp32 accumulation.
//
//  * Stage state z is kept in *A-fragment-swizzled bf16* in global memory, so
//    GEMM1 A-fragments are direct coalesced global_load_b128 (no LDS staging,
//    no per-stage f32->bf16 conversion pass). fp32 z lives in registers.
//  * Spline coefficients for the active knot segment are copied to LDS once
//    per knot change via gfx1250 global_load_async_to_lds_b128 (probe showed
//    the builtin exists; signature is (v4i AS1*, v4i AS3*, i32, i32)).
//    dX/dt is evaluated inline in the contraction.
// ---------------------------------------------------------------------------

typedef __bf16 bf16_t;
typedef __attribute__((ext_vector_type(16))) __bf16 v16bf;
typedef __attribute__((ext_vector_type(8)))  float  v8f;

#define Bn      128
#define Ln      64
#define Cn      32
#define Hn      128
#define HIDn    128
#define OUTn    10
#define NSTEP   126        // (L-1) * 2 RK4 substeps
#define NBLK    32
#define NTHR    256
#define ZS      136        // padded bf16 LDS row stride

// LDS layout (bytes):
//   [0,      34816) : hbuf bf16 [128][136]   (GEMM1 output / GEMM2 A source)
//   [34816,  69632) : fbuf bf16 [128][136]   (GEMM2 output f = tanh(...))
//   [69632, 118784) : cofs f32  [128][96]    (b, 2c, 3d of active segment)
#define LDS_BYTES 118784

#if defined(__HIP_DEVICE_COMPILE__) && __has_builtin(__builtin_amdgcn_global_load_async_to_lds_b128)
#define HAVE_ASYNC_LDS 1
typedef int v4i_t __attribute__((vector_size(16)));
typedef __attribute__((address_space(1))) v4i_t* g_v4i_p;   // global
typedef __attribute__((address_space(3))) v4i_t* l_v4i_p;   // LDS
#endif

__device__ __forceinline__ unsigned bfbits(float x) {
  unsigned u = __float_as_uint(x);
  u += 0x7FFFu + ((u >> 16) & 1u);            // round to nearest even
  return u >> 16;
}
__device__ __forceinline__ bf16_t f2bf(float x) {
  unsigned short s = (unsigned short)bfbits(x);
  return __builtin_bit_cast(bf16_t, s);
}

union FragU { v16bf v; unsigned u[8]; uint4 q[2]; };

// A-matrix (16x32 bf16) fragment gather from LDS, row-major stride ZS.
// ISA layout: lane L holds row M = L%16; lanes 0-15 K={0..7,16..23},
// lanes 16-31 K shifted by +8. VGPR i holds K pair (k0, k0+1).
__device__ __forceinline__ v16bf load_afrag_lds(const bf16_t* lds, int mbase, int kbase) {
  const int lane = threadIdx.x & 31;
  const int half = lane >> 4;
  const int r    = lane & 15;
  const bf16_t* p = lds + (mbase + r) * ZS + kbase + 8 * half;
  FragU a;
#pragma unroll
  for (int i = 0; i < 8; ++i) {
    const int k0 = (i < 4) ? (2 * i) : (16 + 2 * (i - 4));
    a.u[i] = *(const unsigned*)(p + k0);      // ds_load_b32 (fused to 2addr)
  }
  return a.v;
}

// Fragment load from fragment-swizzled global memory:
// layout [tile][ktile][lane][16 bf16] -> two coalesced b128 loads per lane.
__device__ __forceinline__ v16bf load_frag_sw(const bf16_t* sw, int tile, int kt) {
  const int lane = threadIdx.x & 31;
  const uint4* p = (const uint4*)(sw + (((tile * 4) + kt) * 32 + lane) * 16);
  FragU f;
  f.q[0] = p[0];
  f.q[1] = p[1];
  return f.v;
}

__device__ __forceinline__ v8f wmma_bf16(v16bf a, v16bf b, v8f c) {
  return __builtin_amdgcn_wmma_f32_16x16x32_bf16(
      /*neg_a=*/false, a, /*neg_b=*/false, b,
      /*c_mod=*/(short)0, c, /*reuse_a=*/false, /*reuse_b=*/false);
}

// Convert fp32 row-major W[128][Ntot] into WMMA-B fragment-swizzled bf16.
// Lane's 16 values for (ntile,kt): W[kt*32 + (j<8 ? 8*half + j : 16 + 8*half + j - 8)][ntile*16 + r].
__device__ void swizzle_weights(const float* __restrict__ W, bf16_t* __restrict__ Wsw,
                                int ntile0, int ntiles, int Ntot) {
  const int total = ntiles * 4 * 32 * 16;
  for (int e = threadIdx.x; e < total; e += NTHR) {
    const int j    = e & 15;
    const int lane = (e >> 4) & 31;
    const int kt   = (e >> 9) & 3;
    const int nt   = e >> 11;
    const int half = lane >> 4, r = lane & 15;
    const int krow = kt * 32 + ((j < 8) ? (8 * half + j) : (16 + 8 * half + (j - 8)));
    const int ncol = (ntile0 + nt) * 16 + r;
    Wsw[(((size_t)(ntile0 + nt) * 4 + kt) * 32 + lane) * 16 + j] =
        f2bf(W[(size_t)krow * Ntot + ncol]);
  }
}

// Grid-wide sense barrier (all NBLK blocks co-resident: 32 WGs << #WGPs).
__device__ void grid_barrier(unsigned* cnt, unsigned* gen) {
  __threadfence();
  __syncthreads();
  if (threadIdx.x == 0) {
    const unsigned g = __hip_atomic_load(gen, __ATOMIC_ACQUIRE, __HIP_MEMORY_SCOPE_AGENT);
    const unsigned a = __hip_atomic_fetch_add(cnt, 1u, __ATOMIC_ACQ_REL, __HIP_MEMORY_SCOPE_AGENT);
    if (a == NBLK - 1) {
      __hip_atomic_store(cnt, 0u, __ATOMIC_RELAXED, __HIP_MEMORY_SCOPE_AGENT);
      __hip_atomic_fetch_add(gen, 1u, __ATOMIC_RELEASE, __HIP_MEMORY_SCOPE_AGENT);
    } else {
      while (__hip_atomic_load(gen, __ATOMIC_ACQUIRE, __HIP_MEMORY_SCOPE_AGENT) == g) {
        __builtin_amdgcn_s_sleep(2);
      }
    }
  }
  __syncthreads();
}

__device__ __forceinline__ void wait_async_copies() {
#if defined(__HIP_DEVICE_COMPILE__)
#if __has_builtin(__builtin_amdgcn_s_wait_asynccnt)
  __builtin_amdgcn_s_wait_asynccnt(0);
#else
  asm volatile("s_wait_asynccnt 0x0" ::: "memory");
#endif
#endif
}

__global__ void __launch_bounds__(NTHR, 1)
ncde_persistent(const float* __restrict__ coeffs,   // [B][L-1][4C]
                const float* __restrict__ W_init,   // [C][H]
                const float* __restrict__ b_init,   // [H]
                const float* __restrict__ W1,       // [H][HID]
                const float* __restrict__ b1,       // [HID]
                const float* __restrict__ W2,       // [HID][C*H]
                const float* __restrict__ b2,       // [C*H]
                const float* __restrict__ Wr,       // [H][OUT]
                const float* __restrict__ br,       // [OUT]
                float* __restrict__ out,            // [B][OUT]
                unsigned* __restrict__ bar,         // {cnt, gen}
                bf16_t* __restrict__ zsw,           // [8][4][32][16] A-swizzled z
                float* __restrict__ zfin,           // [B][H] final z (readout)
                bf16_t* __restrict__ W1sw,          // [8][4][32][16]
                bf16_t* __restrict__ W2sw)          // [256][4][32][16]
{
  extern __shared__ char smem[];
  bf16_t* hbuf = (bf16_t*)(smem);              // [128][ZS]
  bf16_t* fbuf = (bf16_t*)(smem + 34816);      // [128][ZS]
  float*  cofs = (float*)(smem + 69632);       // [128][96]

  const int blk  = blockIdx.x;                 // owns H columns [4*blk, 4*blk+4)
  const int tid  = threadIdx.x;
  const int wave = tid >> 5;
  const int lane = tid & 31;
  const int half = lane >> 4;
  const int r    = lane & 15;

  // ---- one-time init: weight swizzle (disjoint chunks per block) ----------
  swizzle_weights(W2, W2sw, blk * 8, 8, Cn * Hn);
  if (blk < 8) swizzle_weights(W1, W1sw, blk, 1, HIDn);

  // ---- per-thread RK state: batch row `brow`, two H-columns j0, j0+1 ------
  const int brow = tid >> 1;
  const int j0   = (tid & 1) * 2;
  const int col0 = blk * 4 + j0;               // even; col0,col0+1 owned

  // Precompute this thread's slot in the A-swizzled z layout.
  // k index = col: ktile = col>>5, kk = col&31; half = (kk>>3)&1;
  // k' = kk - 8*half in {0..7,16..23}; dword i; elements 2i, 2i+1.
  const int zkt   = col0 >> 5;
  const int zkk   = col0 & 31;
  const int zhalf = (zkk >> 3) & 1;
  const int zkp   = zkk - 8 * zhalf;
  const int zi    = (zkp < 8) ? (zkp >> 1) : (4 + ((zkp - 16) >> 1));
  const int zlane = (brow & 15) + 16 * zhalf;
  const int zmt   = brow >> 4;
  unsigned* zslot = (unsigned*)(zsw + ((((size_t)zmt * 4 + zkt) * 32 + zlane) * 16 + 2 * zi));

  // ---- one-time init: z0 = a[:,0,:] @ W_init + b_init for owned columns ---
  float zl0, zl1, ka0 = 0.f, ka1 = 0.f;
  {
    const float* ac = coeffs + (size_t)brow * ((Ln - 1) * 4 * Cn);  // 'a', seg 0
    float s0 = b_init[col0], s1 = b_init[col0 + 1];
    for (int c = 0; c < Cn; ++c) {
      const float av = ac[c];
      s0 += av * W_init[c * Hn + col0];
      s1 += av * W_init[c * Hn + col0 + 1];
    }
    zl0 = s0; zl1 = s1;
    *zslot = bfbits(s0) | (bfbits(s1) << 16);
  }
  grid_barrier(bar, bar + 1);

  const float dt = 0.5f;
  const float wst[4] = { dt / 6.f, dt / 3.f, dt / 3.f, dt / 6.f };
  const float cnx[3] = { 0.25f, 0.25f, 0.5f };

  int prev_idx = -1;

  for (int it = 0; it < NSTEP * 4; ++it) {
    const int step = it >> 2;
    const int stg  = it & 3;
    const int tq   = 2 * step + ((stg == 1 || stg == 2) ? 1 : (stg == 3 ? 2 : 0));
    int idx        = tq >> 2; if (idx > Ln - 2) idx = Ln - 2;
    const float fr = 0.25f * (float)tq - (float)idx;

    // ---- refresh LDS spline segment (b,2c,3d: 96 f32 per row) on knot move.
    // Issued right after the barrier so the copy overlaps both GEMMs.
    const bool seg_new = (idx != prev_idx);
    prev_idx = idx;
    if (seg_new) {
      for (int e = tid; e < Bn * 24; e += NTHR) {
        const int b = e / 24, q = e % 24;        // 24 x 16B per row
        const float* gp = coeffs + ((size_t)b * (Ln - 1) + idx) * (4 * Cn) + Cn + q * 4;
        float* lp = cofs + b * 96 + q * 4;
#ifdef HAVE_ASYNC_LDS
        __builtin_amdgcn_global_load_async_to_lds_b128((g_v4i_p)gp, (l_v4i_p)lp, 0, 0);
#else
        *(float4*)lp = *(const float4*)gp;
#endif
      }
    }

    // ---- GEMM1: h = relu(z @ W1 + b1); A-frags straight from global zsw ---
    {
      const int mbase = wave * 16;
      v16bf afr[4];
#pragma unroll
      for (int kt = 0; kt < 4; ++kt) afr[kt] = load_frag_sw(zsw, wave, kt);
#pragma unroll
      for (int nt = 0; nt < 8; ++nt) {
        v8f acc = {0.f, 0.f, 0.f, 0.f, 0.f, 0.f, 0.f, 0.f};
#pragma unroll
        for (int kt = 0; kt < 4; ++kt)
          acc = wmma_bf16(afr[kt], load_frag_sw(W1sw, nt, kt), acc);
        const int colb = nt * 16 + r;
        const float bias = b1[colb];
#pragma unroll
        for (int i = 0; i < 8; ++i) {
          const int row = mbase + i + 8 * half;
          float v = acc[i] + bias;
          hbuf[row * ZS + colb] = f2bf(v > 0.f ? v : 0.f);
        }
      }
    }
    __syncthreads();

    // ---- GEMM2: f = tanh(h @ W2[:, blk*128:+128] + b2) -> fbuf -----------
    {
      const int mbase = wave * 16;
      v16bf afr[4];
#pragma unroll
      for (int kt = 0; kt < 4; ++kt) afr[kt] = load_afrag_lds(hbuf, mbase, kt * 32);
#pragma unroll
      for (int nt = 0; nt < 8; ++nt) {
        v8f acc = {0.f, 0.f, 0.f, 0.f, 0.f, 0.f, 0.f, 0.f};
#pragma unroll
        for (int kt = 0; kt < 4; ++kt)
          acc = wmma_bf16(afr[kt], load_frag_sw(W2sw, blk * 8 + nt, kt), acc);
        const int colb = nt * 16 + r;
        const float bias = b2[blk * 128 + colb];
#pragma unroll
        for (int i = 0; i < 8; ++i) {
          const int row = mbase + i + 8 * half;
          fbuf[row * ZS + colb] = f2bf(tanhf(acc[i] + bias));
        }
      }
    }
#ifdef HAVE_ASYNC_LDS
    if (seg_new) wait_async_copies();
#endif
    __syncthreads();

    // ---- contraction k[b,h] = sum_c f[b,h,c]*dX[b,c] + RK4 update --------
    {
      const float ws = wst[stg];
      const float* cr = cofs + brow * 96;              // b | 2c | 3d
      const bf16_t* frow = fbuf + brow * ZS + j0 * 32; // f for col0 / col0+1
      float k0 = 0.f, k1 = 0.f;
#pragma unroll
      for (int c = 0; c < 32; c += 2) {
        const float dx0 = cr[c]     + fr * (cr[32 + c]     + fr * cr[64 + c]);
        const float dx1 = cr[c + 1] + fr * (cr[32 + c + 1] + fr * cr[64 + c + 1]);
        const unsigned u0 = *(const unsigned*)(frow + c);
        const unsigned u1 = *(const unsigned*)(frow + 32 + c);
        k0 += __uint_as_float(u0 << 16) * dx0 + __uint_as_float(u0 & 0xFFFF0000u) * dx1;
        k1 += __uint_as_float(u1 << 16) * dx0 + __uint_as_float(u1 & 0xFFFF0000u) * dx1;
      }
      ka0 += ws * k0;
      ka1 += ws * k1;
      float zn0, zn1;
      if (stg < 3) {
        zn0 = zl0 + cnx[stg] * k0;
        zn1 = zl1 + cnx[stg] * k1;
      } else {
        zl0 += ka0; zl1 += ka1;
        ka0 = 0.f;  ka1 = 0.f;
        zn0 = zl0;  zn1 = zl1;
      }
      *zslot = bfbits(zn0) | (bfbits(zn1) << 16);      // bf16 A-swizzled state
      if (it == NSTEP * 4 - 1) {                       // fp32 copy for readout
        zfin[brow * Hn + col0]     = zn0;
        zfin[brow * Hn + col0 + 1] = zn1;
      }
    }
    grid_barrier(bar, bar + 1);   // also guards LDS reuse next iteration
  }

  // ---- readout: out = zT @ Wr + br (tiny; block 0, plain VALU) -----------
  if (blk == 0) {
    for (int e = tid; e < Bn * OUTn; e += NTHR) {
      const int b = e / OUTn, o = e % OUTn;
      float s = br[o];
      for (int h = 0; h < Hn; ++h) s += zfin[b * Hn + h] * Wr[h * OUTn + o];
      out[e] = s;
    }
  }
}

extern "C" void kernel_launch(void* const* d_in, const int* in_sizes, int n_in,
                              void* d_out, int out_size, void* d_ws, size_t ws_size,
                              hipStream_t stream) {
  (void)in_sizes; (void)n_in; (void)out_size; (void)ws_size;
  const float* coeffs = (const float*)d_in[0];
  const float* W_init = (const float*)d_in[1];
  const float* b_init = (const float*)d_in[2];
  const float* W1     = (const float*)d_in[3];
  const float* b1     = (const float*)d_in[4];
  const float* W2     = (const float*)d_in[5];
  const float* b2     = (const float*)d_in[6];
  const float* Wr     = (const float*)d_in[7];
  const float* br     = (const float*)d_in[8];
  float* out = (float*)d_out;

  char* ws = (char*)d_ws;
  unsigned* bar  = (unsigned*)ws;                                  // 256 B
  bf16_t* zsw    = (bf16_t*)(ws + 256);                            // 32 KB
  float*  zfin   = (float*)(ws + 256 + 32768);                     // 64 KB
  bf16_t* W1sw   = (bf16_t*)(ws + 256 + 32768 + 65536);            // 32 KB
  bf16_t* W2sw   = (bf16_t*)(ws + 256 + 32768 + 65536 + 32768);    // 1 MB

  (void)hipMemsetAsync(d_ws, 0, 256, stream);                      // barrier state
  ncde_persistent<<<NBLK, NTHR, LDS_BYTES, stream>>>(
      coeffs, W_init, b_init, W1, b1, W2, b2, Wr, br,
      out, bar, zsw, zfin, W1sw, W2sw);
}
